// Linear_54305566490921
// MI455X (gfx1250) — matrix operand order repaired and checked by reading, not compiled, not run on hardware
//
#include <hip/hip_runtime.h>

// ---------------------------------------------------------------------------
// Problem: OUT=2048, IN=2048, K=8, BATCH=256
//   net[o][i] = weight[o][i][argmax_k score[o][i][k]]
//   out[b][o] = sum_i x[b][i] * net[o][i]
// Pass 1: bandwidth-bound select (268 MB streamed once), emits bf16 hi/lo split
// Pass 2: v_wmma_f32_16x16x32_bf16 GEMM with bf16x3 decomposition (~fp32 acc.)
// ---------------------------------------------------------------------------

#define OUT_DIM 2048
#define IN_DIM  2048
#define KCAND   8
#define BATCH   256

typedef __attribute__((ext_vector_type(16))) __bf16 v16bf;
typedef __attribute__((ext_vector_type(8)))  __bf16 v8bf;
typedef __attribute__((ext_vector_type(8)))  float  v8f;
typedef __attribute__((ext_vector_type(4)))  float  v4f;

// ---------------------------------------------------------------------------
// Pass 1a: top-1 select over K=8 candidates + fp32 -> bf16 hi/lo split of net.
// One thread per (o,i) slot; 2x float4 loads of score, 2x float4 of weight.
// Strict '>' keeps the FIRST max, matching jnp.argmax semantics.
// ---------------------------------------------------------------------------
__global__ void __launch_bounds__(256)
select_split_kernel(const float* __restrict__ weight,
                    const float* __restrict__ score,
                    __bf16* __restrict__ netH,
                    __bf16* __restrict__ netL) {
    size_t idx = (size_t)blockIdx.x * blockDim.x + threadIdx.x;  // 0 .. OUT*IN-1
    const v4f* sp = (const v4f*)(score  + idx * KCAND);
    const v4f* wp = (const v4f*)(weight + idx * KCAND);
    v4f s0 = sp[0], s1 = sp[1];
    v4f w0 = wp[0], w1 = wp[1];
    float s[KCAND] = {s0.x, s0.y, s0.z, s0.w, s1.x, s1.y, s1.z, s1.w};
    float w[KCAND] = {w0.x, w0.y, w0.z, w0.w, w1.x, w1.y, w1.z, w1.w};
    float best = s[0];
    float wsel = w[0];
#pragma unroll
    for (int k = 1; k < KCAND; ++k) {
        if (s[k] > best) { best = s[k]; wsel = w[k]; }
    }
    __bf16 h = (__bf16)wsel;                  // RNE round to bf16 (hi part)
    netH[idx] = h;
    netL[idx] = (__bf16)(wsel - (float)h);    // residual (lo part)
}

// ---------------------------------------------------------------------------
// Pass 1b: fp32 -> bf16 hi/lo split of x (256x2048, tiny).
// ---------------------------------------------------------------------------
__global__ void __launch_bounds__(256)
xsplit_kernel(const float* __restrict__ x,
              __bf16* __restrict__ xH,
              __bf16* __restrict__ xL) {
    size_t idx = (size_t)blockIdx.x * blockDim.x + threadIdx.x;
    float v = x[idx];
    __bf16 h = (__bf16)v;
    xH[idx] = h;
    xL[idx] = (__bf16)(v - (float)h);
}

// ---------------------------------------------------------------------------
// Fragment loaders for 16-bit WMMA; A and B share one contiguous k order, k = 16*(lane>>4) + s.
// A (16x32, MxK): lane m=lane&15; VGPR0-3 hold K = (lane>>4)*16 .. +7,
//                 VGPR4-7 hold K = (lane>>4)*16+8 .. +7  -> two 16B loads.
// B (32x16, KxN): lane n=lane&15; lanes 0-15 hold K=0..15, lanes 16-31 hold
//                 K=16..31 -> one contiguous 32B load (net is [o][i] row-major,
//                 so netT's column o is contiguous in i).
// ---------------------------------------------------------------------------
__device__ __forceinline__ v16bf load_a_frag(const __bf16* __restrict__ base,
                                             int lane, int row0, int k0) {
    const int m  = lane & 15;
    const int ks = (lane >> 4) * 16;
    const __bf16* p = base + (size_t)(row0 + m) * IN_DIM + (k0 + ks);
    v8bf lo = *(const v8bf*)p;
    v8bf hi = *(const v8bf*)(p + 8);
    return __builtin_shufflevector(lo, hi, 0, 1, 2, 3, 4, 5, 6, 7,
                                           8, 9, 10, 11, 12, 13, 14, 15);
}

__device__ __forceinline__ v16bf load_b_frag(const __bf16* __restrict__ base,
                                             int lane, int col0, int k0) {
    const int n  = lane & 15;
    const int ks = (lane >> 4) * 16;
    return *(const v16bf*)(base + (size_t)(col0 + n) * IN_DIM + (k0 + ks));
}

__device__ __forceinline__ v8f wmma_bf16(v16bf a, v16bf b, v8f c) {
    // (neg_a, A, neg_b, B, c_mod, C, reuse_a, reuse_b)
    return __builtin_amdgcn_wmma_f32_16x16x32_bf16(false, a, false, b,
                                                   (short)0, c, false, false);
}

// ---------------------------------------------------------------------------
// Pass 2: out[256,2048] = x . netT via bf16x3 split:
//   x.w ~= xH.wH + xH.wL + xL.wH   (f32 accumulate; lo.lo term ~2^-18, dropped)
// Grid: 32 blocks x 256 threads (8 waves). Block b owns output columns
// [64b, 64b+64); wave w owns rows [32w, 32w+32). 24 WMMAs per K-step of 32.
// Full blocks, no divergence -> EXEC all-ones as WMMA requires.
// ---------------------------------------------------------------------------
__global__ void __launch_bounds__(256)
gemm_bf16x3_kernel(const __bf16* __restrict__ xH,
                   const __bf16* __restrict__ xL,
                   const __bf16* __restrict__ netH,
                   const __bf16* __restrict__ netL,
                   float* __restrict__ out) {
    const int lane = threadIdx.x & 31;
    const int wave = threadIdx.x >> 5;
    const int o0   = blockIdx.x * 64;   // output-column tile of this block
    const int b0   = wave * 32;         // batch-row tile of this wave

    const v8f zero = {0.f, 0.f, 0.f, 0.f, 0.f, 0.f, 0.f, 0.f};
    v8f acc[2][4];
#pragma unroll
    for (int mi = 0; mi < 2; ++mi)
#pragma unroll
        for (int ni = 0; ni < 4; ++ni) acc[mi][ni] = zero;

    for (int k0 = 0; k0 < IN_DIM; k0 += 32) {
        // Prefetch next K-tile of the B operand (emits global_prefetch_b8).
        if (k0 + 32 < IN_DIM) {
            __builtin_prefetch(netH + (size_t)(o0 + (lane & 15)) * IN_DIM + k0 + 32, 0, 1);
            __builtin_prefetch(netL + (size_t)(o0 + (lane & 15)) * IN_DIM + k0 + 32, 0, 1);
        }

        v16bf aH0 = load_a_frag(xH, lane, b0,      k0);
        v16bf aH1 = load_a_frag(xH, lane, b0 + 16, k0);
        v16bf aL0 = load_a_frag(xL, lane, b0,      k0);
        v16bf aL1 = load_a_frag(xL, lane, b0 + 16, k0);

#pragma unroll
        for (int ni = 0; ni < 4; ++ni) {
            v16bf bH = load_b_frag(netH, lane, o0 + ni * 16, k0);
            v16bf bL = load_b_frag(netL, lane, o0 + ni * 16, k0);
            acc[0][ni] = wmma_bf16(aH0, bH, acc[0][ni]);
            acc[0][ni] = wmma_bf16(aH0, bL, acc[0][ni]);
            acc[0][ni] = wmma_bf16(aL0, bH, acc[0][ni]);
            acc[1][ni] = wmma_bf16(aH1, bH, acc[1][ni]);
            acc[1][ni] = wmma_bf16(aH1, bL, acc[1][ni]);
            acc[1][ni] = wmma_bf16(aL1, bH, acc[1][ni]);
        }
    }

    // C/D layout: VGPR r -> row (r + 8*(lane>>4)), col (lane&15).
    const int ncol = lane & 15;
    const int mrow = (lane >> 4) * 8;
#pragma unroll
    for (int mi = 0; mi < 2; ++mi) {
#pragma unroll
        for (int ni = 0; ni < 4; ++ni) {
            float* op = out + (size_t)(b0 + mi * 16 + mrow) * OUT_DIM
                            + (o0 + ni * 16 + ncol);
#pragma unroll
            for (int r = 0; r < 8; ++r)
                op[(size_t)r * OUT_DIM] = acc[mi][ni][r];
        }
    }
}

// ---------------------------------------------------------------------------
// Host launcher. d_in order follows setup_inputs(): x, weight, score.
// Workspace layout (18 MB): netH(8MB) | netL(8MB) | xH(1MB) | xL(1MB).
// ---------------------------------------------------------------------------
extern "C" void kernel_launch(void* const* d_in, const int* in_sizes, int n_in,
                              void* d_out, int out_size, void* d_ws, size_t ws_size,
                              hipStream_t stream) {
    (void)in_sizes; (void)n_in; (void)out_size; (void)ws_size;

    const float* x      = (const float*)d_in[0];
    const float* weight = (const float*)d_in[1];
    const float* score  = (const float*)d_in[2];
    float*       out    = (float*)d_out;

    __bf16* netH = (__bf16*)d_ws;
    __bf16* netL = netH + (size_t)OUT_DIM * IN_DIM;
    __bf16* xH   = netL + (size_t)OUT_DIM * IN_DIM;
    __bf16* xL   = xH   + (size_t)BATCH * IN_DIM;

    // Pass 1: stream weight+score once (268 MB, HBM-bound), emit split net.
    select_split_kernel<<<(OUT_DIM * IN_DIM) / 256, 256, 0, stream>>>(
        weight, score, netH, netL);

    // Pass 1b: split x (tiny).
    xsplit_kernel<<<(BATCH * IN_DIM) / 256, 256, 0, stream>>>(x, xH, xL);

    // Pass 2: WMMA GEMM out of L2-resident split operands.
    gemm_bf16x3_kernel<<<OUT_DIM / 64, 256, 0, stream>>>(xH, xL, netH, netL, out);
}
